// EnhancedLinkPredictor_6700148981974
// MI455X (gfx1250) — matrix operand order repaired and checked by reading, not compiled, not run on hardware
//
#include <hip/hip_runtime.h>

// ---------------------------------------------------------------------------
// EnhancedLinkPredictor for MI455X (gfx1250, wave32, WMMA bf16 16x16x32).
// One wave32 owns a 16-edge M-tile.
//  * All 32 gather loads (A) issued as one burst per tile -> max MLP on the
//    L2 round trip, converted to bf16 fragments as they land (v_perm packs).
//  * B operands depth-2 software-pipelined through a rotating register pair,
//    so no wmma waits on a just-issued ds_load.
//  * ks-outer/nt-inner: 8 independent accumulator chains = back-to-back XDL.
// ---------------------------------------------------------------------------

typedef __bf16 bf16_t;
typedef __attribute__((ext_vector_type(16))) bf16_t       v16bf;
typedef __attribute__((ext_vector_type(8)))  float        v8f;
typedef __attribute__((ext_vector_type(8)))  unsigned int u32x8;

constexpr int kH  = 128;   // hidden dim
constexpr int k2H = 256;   // 2*H (layer-1 K)
constexpr int kHQ = 64;    // H/2
constexpr int kWavesPerBlock = 8;   // 256 threads

// RNE f32->bf16 (weight staging only, one-time)
__device__ __forceinline__ unsigned int f2bf_rne(float f) {
  unsigned int u = __builtin_bit_cast(unsigned int, f);
  return (u + 0x7FFFu + ((u >> 16) & 1u)) >> 16;
}
// single v_perm_b32: two f32 -> packed bf16 (truncate) — hot path
__device__ __forceinline__ unsigned int pk2bf(float lo, float hi) {
  return __builtin_amdgcn_perm(__builtin_bit_cast(unsigned int, hi),
                               __builtin_bit_cast(unsigned int, lo),
                               0x07060302u);
}
__device__ __forceinline__ v16bf cast16bf(const u32x8& p) {
  return __builtin_bit_cast(v16bf, p);
}
__device__ __forceinline__ u32x8 pack2q(uint4 x, uint4 y) {
  u32x8 b;
  b[0] = x.x; b[1] = x.y; b[2] = x.z; b[3] = x.w;
  b[4] = y.x; b[5] = y.y; b[6] = y.z; b[7] = y.w;
  return b;
}

__global__ __launch_bounds__(256, 2)
void edge_mlp_wmma(const float* __restrict__ emb,
                   const long long* __restrict__ eidx,   // [2][E]
                   const float* __restrict__ W1, const float* __restrict__ b1,
                   const float* __restrict__ W2, const float* __restrict__ b2,
                   const float* __restrict__ W3, const float* __restrict__ b3,
                   float* __restrict__ out, int E, int numTiles)
{
  __shared__ __align__(16) unsigned short lds_w1t[kH][k2H];   // 64 KB
  __shared__ __align__(16) unsigned short lds_w2t[kHQ][kH];   // 16 KB
  __shared__ float lds_b1[kH];
  __shared__ float lds_b2[kHQ];
  __shared__ __align__(16) float lds_w3[kHQ];
  __shared__ float lds_b3s;
  __shared__ __align__(16) unsigned short lds_h1[kWavesPerBlock][16][kH];  // 32 KB
  __shared__ __align__(16) float          lds_h2[kWavesPerBlock][16][kHQ]; // 32 KB

  const int tid = threadIdx.x;

  // ---- stage weights (transpose + RNE fp32->bf16 + swizzle), once --------
  for (int i = tid; i < k2H * kH; i += 256) {          // W1[k][n]
    int k = i >> 7, n = i & 127;
    lds_w1t[n][(((k >> 3) + n) & 31) * 8 + (k & 7)] = (unsigned short)f2bf_rne(W1[i]);
  }
  for (int i = tid; i < kH * kHQ; i += 256) {          // W2[k][n]
    int k = i >> 6, n = i & 63;
    lds_w2t[n][(((k >> 3) + n) & 15) * 8 + (k & 7)] = (unsigned short)f2bf_rne(W2[i]);
  }
  if (tid < kH)  lds_b1[tid] = b1[tid];
  if (tid < kHQ) { lds_b2[tid] = b2[tid]; lds_w3[tid] = W3[tid]; }
  if (tid == 0)  lds_b3s = b3[0];
  __syncthreads();

  const int lane = tid & 31;
  const int wave = tid >> 5;
  const int m    = lane & 15;   // M row (edge) / N within a 16-tile
  const int h    = lane >> 4;   // lane half

  const int waveGlobal = blockIdx.x * kWavesPerBlock + wave;
  const int waveStride = gridDim.x * kWavesPerBlock;
  const v8f zeroc = {};

  // B-fragment loaders (swizzled, conflict-free ds_load_b128 pairs)
  auto loadB1 = [&](int ks, int nt) -> u32x8 {
    const int b0 = ks * 4 + h;
    const int n  = nt * 16 + m;
    uint4 x = *(const uint4*)&lds_w1t[n][((b0 + n) & 31) * 8];
    uint4 y = *(const uint4*)&lds_w1t[n][((b0 + 2 + n) & 31) * 8];
    return pack2q(x, y);
  };
  auto loadB2 = [&](int ks, int nt) -> u32x8 {
    const int b0 = ks * 4 + h;
    const int n  = nt * 16 + m;
    uint4 x = *(const uint4*)&lds_w2t[n][((b0 + n) & 15) * 8];
    uint4 y = *(const uint4*)&lds_w2t[n][((b0 + 2 + n) & 15) * 8];
    return pack2q(x, y);
  };

  for (int t = waveGlobal; t < numTiles; t += waveStride) {
    const int e  = t * 16 + m;
    const int ec = (e < E) ? e : (E - 1);
    const long long sIdx = eidx[ec];
    const long long dIdx = eidx[(long long)E + ec];
    const float* rowS = emb + sIdx * kH + 8 * h;
    const float* rowD = emb + dIdx * kH + 8 * h;

    // prefetch next tile's gather rows (4 ops cover 8 cachelines)
    const int tn = t + waveStride;
    if (tn < numTiles) {
      int en  = tn * 16 + m;
      int ecn = (en < E) ? en : (E - 1);
      const long long sN = eidx[ecn];
      const long long dN = eidx[(long long)E + ecn];
      const float* pS = emb + sN * kH + h * 64;
      const float* pD = emb + dN * kH + h * 64;
      __builtin_prefetch(pS, 0, 3);
      __builtin_prefetch(pS + 32, 0, 3);
      __builtin_prefetch(pD, 0, 3);
      __builtin_prefetch(pD + 32, 0, 3);
    }

    // ---- A: burst all 32 gather loads, then pack as data lands -----------
    float4 q[32];
    #pragma unroll
    for (int i = 0; i < 32; ++i) {
      const int ks = i >> 2, c = i & 3;
      const float* p = ((ks < 4) ? rowS : rowD)
                       + (ks & 3) * 32 + (c >> 1) * 16 + (c & 1) * 4;
      q[i] = *(const float4*)p;
    }
    u32x8 afr[8];
    #pragma unroll
    for (int ks = 0; ks < 8; ++ks) {
      const float4 f0 = q[4 * ks + 0], f1 = q[4 * ks + 1];
      const float4 f2 = q[4 * ks + 2], f3 = q[4 * ks + 3];
      afr[ks][0] = pk2bf(f0.x, f0.y); afr[ks][1] = pk2bf(f0.z, f0.w);
      afr[ks][2] = pk2bf(f1.x, f1.y); afr[ks][3] = pk2bf(f1.z, f1.w);
      afr[ks][4] = pk2bf(f2.x, f2.y); afr[ks][5] = pk2bf(f2.z, f2.w);
      afr[ks][6] = pk2bf(f3.x, f3.y); afr[ks][7] = pk2bf(f3.z, f3.w);
    }

    // ================= layer 1: [16,256] @ [256,128] ======================
    v8f acc[8];
    {
      u32x8 bp0 = loadB1(0, 0);
      u32x8 bp1 = loadB1(0, 1);
      #pragma unroll
      for (int i = 0; i < 64; ++i) {                   // i = ks*8 + nt
        const int ks = i >> 3, nt = i & 7;
        u32x8 bnx = {};
        if (i < 62) bnx = loadB1((i + 2) >> 3, (i + 2) & 7);
        const v8f cin = (ks == 0) ? zeroc : acc[nt];
        acc[nt] = __builtin_amdgcn_wmma_f32_16x16x32_bf16(
            false, cast16bf(afr[ks]), false, cast16bf(bp0), (short)0,
            cin, false, false);
        bp0 = bp1; bp1 = bnx;
      }
    }
    // bias + relu + truncate to bf16 + transposed (swizzled) store
    #pragma unroll
    for (int nt = 0; nt < 8; ++nt) {
      const int n = nt * 16 + m;
      const float bias = lds_b1[n];
      const int blk = n >> 3;
      #pragma unroll
      for (int r = 0; r < 8; ++r) {
        const int row = r + 8 * h;
        float v = acc[nt][r] + bias;
        v = v > 0.f ? v : 0.f;
        lds_h1[wave][row][((blk + row) & 15) * 8 + (n & 7)] =
            (unsigned short)(__builtin_bit_cast(unsigned int, v) >> 16);
      }
    }

    // ================= layer 2: [16,128] @ [128,64] =======================
    u32x8 afr2[4];
    #pragma unroll
    for (int ks = 0; ks < 4; ++ks) {
      const int ca = ks * 4 + h;
      uint4 ax = *(const uint4*)&lds_h1[wave][m][((ca + m) & 15) * 8];
      uint4 ay = *(const uint4*)&lds_h1[wave][m][((ca + 2 + m) & 15) * 8];
      afr2[ks] = pack2q(ax, ay);
    }
    v8f acc2[4];
    {
      u32x8 bp0 = loadB2(0, 0);
      u32x8 bp1 = loadB2(0, 1);
      #pragma unroll
      for (int i = 0; i < 16; ++i) {                   // i = ks*4 + nt
        const int ks = i >> 2, nt = i & 3;
        u32x8 bnx = {};
        if (i < 14) bnx = loadB2((i + 2) >> 2, (i + 2) & 3);
        const v8f cin = (ks == 0) ? zeroc : acc2[nt];
        acc2[nt] = __builtin_amdgcn_wmma_f32_16x16x32_bf16(
            false, cast16bf(afr2[ks]), false, cast16bf(bp0), (short)0,
            cin, false, false);
        bp0 = bp1; bp1 = bnx;
      }
    }
    #pragma unroll
    for (int nt = 0; nt < 4; ++nt) {
      const int n = nt * 16 + m;
      const float bias = lds_b2[n];
      #pragma unroll
      for (int r = 0; r < 8; ++r) {
        const int row = r + 8 * h;
        float v = acc2[nt][r] + bias;
        v = v > 0.f ? v : 0.f;
        lds_h2[wave][row][(n + 4 * row) & 63] = v;     // swizzled f32 tile
      }
    }

    // ================= layer 3: [16,64] @ [64,1] ==========================
    {
      float sum = 0.f;
      #pragma unroll
      for (int kq = 0; kq < 8; ++kq) {                 // each half sums 32 terms
        const int kk = kq * 4 + 32 * h;
        const float4 hv = *(const float4*)&lds_h2[wave][m][(kk + 4 * m) & 63];
        const float4 wv = *(const float4*)&lds_w3[kk];
        sum += hv.x * wv.x + hv.y * wv.y + hv.z * wv.z + hv.w * wv.w;
      }
      sum += __shfl_xor(sum, 16, 32);                  // combine halves (wave32)
      if (h == 0 && e < E) out[e] = sum + lds_b3s;
    }
  }
}

extern "C" void kernel_launch(void* const* d_in, const int* in_sizes, int n_in,
                              void* d_out, int out_size, void* d_ws, size_t ws_size,
                              hipStream_t stream) {
  const float*     emb  = (const float*)d_in[0];
  const long long* eidx = (const long long*)d_in[1];
  const float*     W1   = (const float*)d_in[2];
  const float*     b1   = (const float*)d_in[3];
  const float*     W2   = (const float*)d_in[4];
  const float*     b2   = (const float*)d_in[5];
  const float*     W3   = (const float*)d_in[6];
  const float*     b3   = (const float*)d_in[7];
  float* out = (float*)d_out;

  const int E = in_sizes[1] / 2;            // edge_index is [2, E] int64
  const int numTiles = (E + 15) / 16;
  int blocks = (numTiles + kWavesPerBlock - 1) / kWavesPerBlock;
  if (blocks > 512) blocks = 512;           // persistent: amortize weight staging

  edge_mlp_wmma<<<dim3(blocks), dim3(256), 0, stream>>>(
      emb, eidx, W1, b1, W2, b2, W3, b3, out, E, numTiles);
}
